// NeuralTensorLayer_7756710936975
// MI455X (gfx1250) — matrix operand order repaired and checked by reading, not compiled, not run on hardware
//
#include <hip/hip_runtime.h>

// ---------------------------------------------------------------------------
// Neural Tensor Layer on MI455X (gfx1250), wave32.
//   ntn(a,c,T)[b,k] = tanh( sum_ij a[b,i]T[k,i,j]c[b,j] + W@[a;c] + bias )
// One GEMM per ntn: (B x 4224)x(4224 x 64); A-rows are the outer product
// a(x)c synthesized in registers (v_pk_mul_bf16 on cached packed-bf16
// c-slices), B is a bf16 panel pre-swizzled to the WMMA B-fragment layout.
// The panel is staged to LDS once per workgroup with double-buffered
// global_load_async_to_lds_b128 (ASYNCcnt), consumed via ds_load_b128,
// accumulated with v_wmma_f32_16x16x32_bf16 (2 M-tiles x 4 N-tiles / wave).
// ---------------------------------------------------------------------------

typedef __attribute__((ext_vector_type(16))) __bf16 v16bf;
typedef __attribute__((ext_vector_type(8)))  float  v8f;

#define BATCH        32768
#define NCHUNK       132                      // K = 132*32 = 4224
#define PANEL_ELEMS  (NCHUNK * 4 * 32 * 16)   // 270336 bf16 per panel
#define PANEL_STRIDE (1u << 20)               // 1 MB spacing in workspace
#define GROUP_BYTES  16384                    // 4 chunks * 4KB staged per step

// A-matrix (16-bit, 16x32) layout: element e, lane-half h -> K index.
__device__ __forceinline__ int a_klane(int e, int h) {
    int v = e >> 1;
    return ((v >> 2) << 4) + (h << 3) + ((v & 3) << 1) + (e & 1);
}

__device__ __forceinline__ uint32_t pack_bf16(float lo, float hi) {
    unsigned short l = __builtin_bit_cast(unsigned short, (__bf16)lo);
    unsigned short h = __builtin_bit_cast(unsigned short, (__bf16)hi);
    return (uint32_t)l | ((uint32_t)h << 16);
}

__device__ __forceinline__ uint32_t pkmul_bf16(uint32_t a, uint32_t b) {
    uint32_t d;
    asm("v_pk_mul_bf16 %0, %1, %2" : "=v"(d) : "v"(a), "v"(b));
    return d;
}

union FragU {
    uint32_t u[8];
    v16bf    v;
};

// Build swizzled bf16 B-panels: panel[d], d = ((t*4+nt)*32+lane)*16+e,
// K = t*32 + (e/8)*16 + (lane/16)*8 + e%8, N = nt*16 + lane%16.
__global__ void prep_panels(const float* __restrict__ T1,
                            const float* __restrict__ T2,
                            const float* __restrict__ T3,
                            const float* __restrict__ W,
                            __bf16* __restrict__ P1,
                            __bf16* __restrict__ P2,
                            __bf16* __restrict__ P3) {
    int idx = blockIdx.x * blockDim.x + threadIdx.x;
    if (idx >= 3 * PANEL_ELEMS) return;
    int panel = idx / PANEL_ELEMS;
    int d = idx - panel * PANEL_ELEMS;

    const float* T = (panel == 0) ? T1 : (panel == 1) ? T2 : T3;
    __bf16*      P = (panel == 0) ? P1 : (panel == 1) ? P2 : P3;

    int e    = d & 15;
    int lane = (d >> 4) & 31;
    int nt   = (d >> 9) & 3;
    int t    = d >> 11;

    int klB = ((e >> 3) << 4) + ((lane >> 4) << 3) + (e & 7);
    int kk  = t * 32 + klB;
    int n   = nt * 16 + (lane & 15);

    float v;
    if (kk < 4096)      v = T[n * 4096 + kk];              // Tflat[(i,j), n]
    else if (kk < 4160) v = W[n * 128 + (kk - 4096)];      // W[:, 0:64]
    else                v = W[n * 128 + 64 + (kk - 4160)]; // W[:, 64:128]
    P[d] = (__bf16)v;
}

// 8 waves x 32 rows = 256 batch rows per block; grid = BATCH/256.
__global__ __launch_bounds__(256) void ntn_kernel(
    const float* __restrict__ a, const float* __restrict__ c,
    const __bf16* __restrict__ panel, const float* __restrict__ bias,
    float* __restrict__ out, int out_rs, int out_cs) {

    __shared__ __align__(128) __bf16 smem[2 * (GROUP_BYTES / 2)];

    const int tid     = threadIdx.x;
    const int lane    = tid & 31;
    const int wave    = tid >> 5;
    const int half    = lane >> 4;
    const int lm      = lane & 15;
    const int rowBase = blockIdx.x * 256 + wave * 32;
    const int row0    = rowBase + lm;        // M-tile 0 row for this lane
    const int row1    = rowBase + 16 + lm;   // M-tile 1 row

    // Packed-bf16 c-slices in A-fragment ordering (pair j covers elems 2j,2j+1)
    uint32_t c0lo[8], c0hi[8], c1lo[8], c1hi[8];
#pragma unroll
    for (int j = 0; j < 8; ++j) {
        int k0 = a_klane(2 * j, half);
        int k1 = a_klane(2 * j + 1, half);
        c0lo[j] = pack_bf16(c[row0 * 64 + k0],      c[row0 * 64 + k1]);
        c0hi[j] = pack_bf16(c[row0 * 64 + 32 + k0], c[row0 * 64 + 32 + k1]);
        c1lo[j] = pack_bf16(c[row1 * 64 + k0],      c[row1 * 64 + k1]);
        c1hi[j] = pack_bf16(c[row1 * 64 + 32 + k0], c[row1 * 64 + 32 + k1]);
    }

    v8f acc[2][4];
#pragma unroll
    for (int m = 0; m < 2; ++m)
#pragma unroll
        for (int nt = 0; nt < 4; ++nt) acc[m][nt] = v8f{};

    // Async-stage one 4-chunk group (16KB) into LDS buffer buf (0/1).
    auto stage = [&](int g, int buf) {
        uint64_t ga = (uint64_t)(uintptr_t)((const char*)panel +
                                            (size_t)g * GROUP_BYTES + tid * 64);
        uint32_t la = (uint32_t)(uintptr_t)&smem[0] +
                      (uint32_t)buf * GROUP_BYTES + (uint32_t)tid * 64;
#pragma unroll
        for (int q = 0; q < 4; ++q) {
            asm volatile("global_load_async_to_lds_b128 %0, %1, off"
                         :: "v"(la + q * 16), "v"(ga + q * 16) : "memory");
        }
    };

    auto doChunkPair = [&](const __bf16* pc, v16bf af0, v16bf af1) {
#pragma unroll
        for (int nt = 0; nt < 4; ++nt) {
            v16bf bf = *(const v16bf*)(pc + nt * 512 + lane * 16);
            acc[0][nt] = __builtin_amdgcn_wmma_f32_16x16x32_bf16(
                false, af0, false, bf, (short)0, acc[0][nt], false, false);
            acc[1][nt] = __builtin_amdgcn_wmma_f32_16x16x32_bf16(
                false, af1, false, bf, (short)0, acc[1][nt], false, false);
        }
    };

    stage(0, 0);
    asm volatile("s_wait_asynccnt 0x0" ::: "memory");
    __syncthreads();

    uint32_t ai0 = 0, ai1 = 0;
#pragma unroll 1
    for (int g = 0; g < 33; ++g) {
        if (g + 1 < 33) stage(g + 1, (g + 1) & 1);
        const __bf16* sb = &smem[(g & 1) * (GROUP_BYTES / 2)];

        if (g < 32) {
#pragma unroll
            for (int cc = 0; cc < 4; ++cc) {
                // chunk t = 4g+cc covers kk = i*64 + (t&1)*32 + [0,32), i=t>>1
                if ((cc & 1) == 0) {
                    int i = (4 * g + cc) >> 1;
                    float f0 = a[row0 * 64 + i];
                    float f1 = a[row1 * 64 + i];
                    ai0 = pack_bf16(f0, f0);
                    ai1 = pack_bf16(f1, f1);
                }
                FragU f0, f1;
                if ((cc & 1) == 0) {
#pragma unroll
                    for (int j = 0; j < 8; ++j) {
                        f0.u[j] = pkmul_bf16(ai0, c0lo[j]);
                        f1.u[j] = pkmul_bf16(ai1, c1lo[j]);
                    }
                } else {
#pragma unroll
                    for (int j = 0; j < 8; ++j) {
                        f0.u[j] = pkmul_bf16(ai0, c0hi[j]);
                        f1.u[j] = pkmul_bf16(ai1, c1hi[j]);
                    }
                }
                doChunkPair(sb + cc * 2048, f0.v, f1.v);
            }
        } else {
            // Tail (ff term): chunks 128/129 = raw a cols 0-31 / 32-63,
            // chunks 130/131 = raw c cols (already cached packed).
            FragU f0, f1;
#pragma unroll
            for (int j = 0; j < 8; ++j) {
                int k0 = a_klane(2 * j, half), k1 = a_klane(2 * j + 1, half);
                f0.u[j] = pack_bf16(a[row0 * 64 + k0], a[row0 * 64 + k1]);
                f1.u[j] = pack_bf16(a[row1 * 64 + k0], a[row1 * 64 + k1]);
            }
            doChunkPair(sb + 0 * 2048, f0.v, f1.v);
#pragma unroll
            for (int j = 0; j < 8; ++j) {
                int k0 = a_klane(2 * j, half), k1 = a_klane(2 * j + 1, half);
                f0.u[j] = pack_bf16(a[row0 * 64 + 32 + k0], a[row0 * 64 + 32 + k1]);
                f1.u[j] = pack_bf16(a[row1 * 64 + 32 + k0], a[row1 * 64 + 32 + k1]);
            }
            doChunkPair(sb + 1 * 2048, f0.v, f1.v);
#pragma unroll
            for (int j = 0; j < 8; ++j) { f0.u[j] = c0lo[j]; f1.u[j] = c1lo[j]; }
            doChunkPair(sb + 2 * 2048, f0.v, f1.v);
#pragma unroll
            for (int j = 0; j < 8; ++j) { f0.u[j] = c0hi[j]; f1.u[j] = c1hi[j]; }
            doChunkPair(sb + 3 * 2048, f0.v, f1.v);
        }

        asm volatile("s_wait_asynccnt 0x0" ::: "memory");
        __syncthreads();
    }

    // Epilogue: +bias, tanh, strided store. C/D layout: M = r + 8*half, N = lm.
    float bn[4];
#pragma unroll
    for (int nt = 0; nt < 4; ++nt) bn[nt] = bias[nt * 16 + lm];
#pragma unroll
    for (int m = 0; m < 2; ++m) {
#pragma unroll
        for (int r = 0; r < 8; ++r) {
            int row = rowBase + m * 16 + r + half * 8;
            size_t ro = (size_t)row * out_rs;
#pragma unroll
            for (int nt = 0; nt < 4; ++nt) {
                out[ro + (size_t)(nt * 16 + lm) * out_cs] =
                    tanhf(acc[m][nt][r] + bn[nt]);
            }
        }
    }
}

extern "C" void kernel_launch(void* const* d_in, const int* in_sizes, int n_in,
                              void* d_out, int out_size, void* d_ws, size_t ws_size,
                              hipStream_t stream) {
    (void)in_sizes; (void)n_in; (void)out_size; (void)ws_size;
    const float* e1 = (const float*)d_in[0];
    const float* p  = (const float*)d_in[1];
    const float* e2 = (const float*)d_in[2];
    const float* er = (const float*)d_in[3];
    const float* T1 = (const float*)d_in[4];
    const float* T2 = (const float*)d_in[5];
    const float* T3 = (const float*)d_in[6];
    const float* W  = (const float*)d_in[7];
    const float* bv = (const float*)d_in[8];

    char* ws = (char*)d_ws;
    __bf16* P1 = (__bf16*)(ws + 0 * (size_t)PANEL_STRIDE);
    __bf16* P2 = (__bf16*)(ws + 1 * (size_t)PANEL_STRIDE);
    __bf16* P3 = (__bf16*)(ws + 2 * (size_t)PANEL_STRIDE);
    float* a1 = (float*)(ws + 3 * (size_t)PANEL_STRIDE);                 // 8 MB
    float* c2 = (float*)(ws + 3 * (size_t)PANEL_STRIDE + (8u << 20));    // 8 MB
    float* ar = (float*)(ws + 3 * (size_t)PANEL_STRIDE + (16u << 20));   // 8 MB

    int prepN = 3 * PANEL_ELEMS;
    prep_panels<<<(prepN + 255) / 256, 256, 0, stream>>>(T1, T2, T3, W, P1, P2, P3);

    dim3 grid(BATCH / 256), block(256);
    float* out = (float*)d_out;
    ntn_kernel<<<grid, block, 0, stream>>>(e1, p, P1, bv, a1, 64, 1);
    ntn_kernel<<<grid, block, 0, stream>>>(p, e2, P2, bv, c2, 64, 1);
    ntn_kernel<<<grid, block, 0, stream>>>(er, p, P1, bv, ar, 64, 1);
    ntn_kernel<<<grid, block, 0, stream>>>(a1, c2, P3, bv, out + 0, 128, 2);
    ntn_kernel<<<grid, block, 0, stream>>>(ar, c2, P3, bv, out + 1, 128, 2);
}
